// MaskRCNN_51883204935894
// MI455X (gfx1250) — matrix operand order repaired and checked by reading, not compiled, not run on hardware
//
#include <hip/hip_runtime.h>
#include <hip/hip_bf16.h>

typedef __attribute__((ext_vector_type(16))) __bf16 bf16x16;
typedef __attribute__((ext_vector_type(8)))  __bf16 bf16x8;
typedef __attribute__((ext_vector_type(8)))  float  v8f;
typedef __attribute__((ext_vector_type(4)))  unsigned int u32x4;
typedef __attribute__((ext_vector_type(8)))  int i32x8;
typedef __attribute__((ext_vector_type(4)))  int i32x4;

static __device__ inline __bf16 f2bf(float f) {
    union { float f; unsigned u; } a; a.f = f;
    unsigned r = (a.u + 0x7fffu + ((a.u >> 16) & 1u)) >> 16;  // RNE
    union { unsigned short s; __bf16 b; } o; o.s = (unsigned short)r;
    return o.b;
}
static __device__ inline unsigned short bfu(__bf16 b) {
    union { __bf16 b; unsigned short s; } u; u.b = b; return u.s;
}

// ---------------- elementwise helpers ----------------
__global__ void cvt_bf16_kernel(const float* __restrict__ src, __bf16* __restrict__ dst, long long n) {
    long long i = (long long)blockIdx.x * blockDim.x + threadIdx.x;
    if (i < n) dst[i] = f2bf(src[i]);
}

__global__ void zero_kernel(float* __restrict__ p, int n) {
    int i = blockIdx.x * blockDim.x + threadIdx.x;
    if (i < n) p[i] = 0.f;
}

// Swizzle w_fc1 [K=100352][N=1024] fp32 -> bf16 in WMMA B-fragment-major order:
// Bp[((nt*KT + kt)*32 + lane)*16 + i] = W[(kt*32 + kb(lane) + i)*N + nt*16 + (lane&15)]
__global__ void swizzleB_fc1_kernel(const float* __restrict__ W, __bf16* __restrict__ Bp) {
    const int KT = 3136;   // 100352/32
    long long e = (long long)blockIdx.x * blockDim.x + threadIdx.x;
    if (e >= (long long)64 * KT * 512) return;
    int i    = (int)(e & 15);
    int lane = (int)((e >> 4) & 31);
    long long frag = e >> 9;
    int kt = (int)(frag % KT);
    int nt = (int)(frag / KT);
    int kb = (lane < 16) ? 0 : 16;
    int k = kt * 32 + kb + i;
    int n = nt * 16 + (lane & 15);
    Bp[e] = f2bf(W[(size_t)k * 1024 + n]);
}

// ---------------- RoIAlign (1 sample/bin) -> pooled bf16 [M][c*196+y*14+x] ----------------
__global__ void roi_pool_kernel(const float* __restrict__ p2, const float* __restrict__ p3,
                                const float* __restrict__ p4, const float* __restrict__ p5,
                                const float* __restrict__ boxes, const int* __restrict__ bidx,
                                const int* __restrict__ imh, const int* __restrict__ imw,
                                __bf16* __restrict__ pooled) {
    long long gid = (long long)blockIdx.x * blockDim.x + threadIdx.x;
    const long long TOT = (long long)512 * 512 * 196;
    if (gid >= TOT) return;
    int m   = (int)(gid / (512 * 196));
    int rem = (int)(gid % (512 * 196));
    int c = rem / 196;
    int p = rem % 196;
    int py = p / 14, px = p % 14;

    float bx1 = boxes[m * 4 + 0], by1 = boxes[m * 4 + 1];
    float bx2 = boxes[m * 4 + 2], by2 = boxes[m * 4 + 3];
    float alpha = (224.0f / 800.0f) * (float)min(imh[0], imw[0]);
    float bw = fabsf(bx2 - bx1), bh = fabsf(by2 - by1);
    float s  = sqrtf(fmaxf(bw * bh, 1e-6f));
    float kf = floorf(4.0f + log2f(s / alpha));
    float lc = fminf(fmaxf(kf - 2.0f, 0.f), 3.f);
    int lvl = (int)lc;

    const float* feat; int H, W; float scale;
    if      (lvl == 0) { feat = p2; H = 200; W = 200; scale = 0.25f;    }
    else if (lvl == 1) { feat = p3; H = 100; W = 100; scale = 0.125f;   }
    else if (lvl == 2) { feat = p4; H = 50;  W = 50;  scale = 0.0625f;  }
    else               { feat = p5; H = 25;  W = 25;  scale = 0.03125f; }

    float x1 = bx1 * scale, y1 = by1 * scale, x2 = bx2 * scale, y2 = by2 * scale;
    float gx = ((float)px + 0.5f) / 14.f, gy = ((float)py + 0.5f) / 14.f;
    float xs = x1 + gx * (x2 - x1);
    float ys = y1 + gy * (y2 - y1);
    float x0f = floorf(xs), y0f = floorf(ys);
    float lx = xs - x0f, ly = ys - y0f;
    int x0  = (int)fminf(fmaxf(x0f,       0.f), (float)(W - 1));
    int x1i = (int)fminf(fmaxf(x0f + 1.f, 0.f), (float)(W - 1));
    int y0  = (int)fminf(fmaxf(y0f,       0.f), (float)(H - 1));
    int y1i = (int)fminf(fmaxf(y0f + 1.f, 0.f), (float)(H - 1));
    int n = bidx[m];
    const float* fb = feat + ((size_t)n * 512 + c) * (size_t)H * W;
    float v00 = fb[y0  * W + x0], v01 = fb[y0  * W + x1i];
    float v10 = fb[y1i * W + x0], v11 = fb[y1i * W + x1i];
    float val = v00 * (1.f - ly) * (1.f - lx) + v01 * (1.f - ly) * lx
              + v10 * ly * (1.f - lx)         + v11 * ly * lx;
    pooled[(size_t)m * 100352 + (size_t)c * 196 + p] = f2bf(val);
}

// ---------------- FC1: [512 x 100352] x [100352 x 1024] bf16 WMMA ----------------
// Each wave: 1 m-tile x 4 n-tiles, B pre-swizzled fragment-major (contiguous b128 loads).
__global__ void __launch_bounds__(256) fc1_wmma_kernel(const __bf16* __restrict__ A,
                                                       const __bf16* __restrict__ Bp,
                                                       const float* __restrict__ bias,
                                                       float* __restrict__ Hout) {
    const int K = 100352, N = 1024, KT = 3136;
    int lane = threadIdx.x & 31;
    int wave = threadIdx.x >> 5;
    int t  = blockIdx.x * 8 + wave;          // 512 wave-jobs
    int mt = t >> 4;                          // 0..31
    int ng = t & 15;                          // n-tile group of 4
    int m0 = mt * 16;

    // A fragment: lanes 0-15 hold K {0..15}; lanes 16-31 hold K {16..31}
    const __bf16* Arow = A + (size_t)(m0 + (lane & 15)) * K + ((lane < 16) ? 0 : 16);
    const __bf16* B0 = Bp + ((size_t)(ng * 4 + 0) * KT) * 512 + lane * 16;
    const __bf16* B1 = Bp + ((size_t)(ng * 4 + 1) * KT) * 512 + lane * 16;
    const __bf16* B2 = Bp + ((size_t)(ng * 4 + 2) * KT) * 512 + lane * 16;
    const __bf16* B3 = Bp + ((size_t)(ng * 4 + 3) * KT) * 512 + lane * 16;

    v8f acc0 = {}, acc1 = {}, acc2 = {}, acc3 = {};
    for (int kt = 0; kt < KT; ++kt) {
        int k0 = kt * 32;
        __builtin_prefetch(Arow + k0 + 512, 0, 3);   // global_prefetch_b8
        bf16x8 alo = *(const bf16x8*)(Arow + k0);
        bf16x8 ahi = *(const bf16x8*)(Arow + k0 + 8);
        bf16x16 a = __builtin_shufflevector(alo, ahi, 0,1,2,3,4,5,6,7,8,9,10,11,12,13,14,15);
        bf16x16 b0 = *(const bf16x16*)(B0 + (size_t)kt * 512);
        bf16x16 b1 = *(const bf16x16*)(B1 + (size_t)kt * 512);
        bf16x16 b2 = *(const bf16x16*)(B2 + (size_t)kt * 512);
        bf16x16 b3 = *(const bf16x16*)(B3 + (size_t)kt * 512);
        acc0 = __builtin_amdgcn_wmma_f32_16x16x32_bf16(false, a, false, b0, (short)0, acc0, false, false);
        acc1 = __builtin_amdgcn_wmma_f32_16x16x32_bf16(false, a, false, b1, (short)0, acc1, false, false);
        acc2 = __builtin_amdgcn_wmma_f32_16x16x32_bf16(false, a, false, b2, (short)0, acc2, false, false);
        acc3 = __builtin_amdgcn_wmma_f32_16x16x32_bf16(false, a, false, b3, (short)0, acc3, false, false);
    }
    int col = lane & 15, rb = (lane < 16) ? 0 : 8;
#pragma unroll
    for (int j = 0; j < 4; j++) {
        v8f acc = (j == 0) ? acc0 : (j == 1) ? acc1 : (j == 2) ? acc2 : acc3;
        int n0 = (ng * 4 + j) * 16;
#pragma unroll
        for (int r = 0; r < 8; r++) {
            float v = acc[r] + bias[n0 + col];
            Hout[(size_t)(m0 + rb + r) * N + n0 + col] = fmaxf(v, 0.f);
        }
    }
}

// ---------------- conv 3x3 as implicit GEMM per box ----------------
// A (weights) tile staged in LDS via TDM tensor_load_to_lds; B (im2col) built
// fragment-major in LDS; fused bias+ReLU+mean epilogue via f32 atomics.
__global__ void __launch_bounds__(256) conv_wmma_kernel(const __bf16* __restrict__ Wc,     // [256][4608]
                                                        const __bf16* __restrict__ pooled, // [512][512*196]
                                                        const float* __restrict__ bconv,
                                                        float* __restrict__ zmean) {       // [512][256]
    const int KC = 4608, P = 196;
    __shared__ alignas(32) unsigned short bt[32 * 16];    // B tile, fragment-major [lane][elem]
    __shared__ alignas(64) unsigned short at[256 * 32];   // A tile [row][k] (16 KB), filled by TDM
    int box = blockIdx.x / 13;
    int ntl = blockIdx.x % 13;
    int n0  = ntl * 16;
    int tid = threadIdx.x;
    int lane = tid & 31, wave = tid >> 5;
    const __bf16* Pb = pooled + (size_t)box * 100352;

    int kgrp = (lane < 16) ? 0 : 16;
    int row0 = wave * 16 + (lane & 15);          // m-tile = wave
    int row1 = (wave + 8) * 16 + (lane & 15);    // m-tile = wave+8

    v8f acc0 = {}, acc1 = {};
    for (int k0 = 0; k0 < KC; k0 += 32) {
        // ---- im2col B tile, fragment-major: bt[L*16+i] = im2col(k0+kb(L)+i, n0+(L&15)) ----
#pragma unroll
        for (int half = 0; half < 2; ++half) {
            int e = tid + half * 256;            // 512 entries, 2 per thread
            int L = e >> 4, i = e & 15;
            int krow = k0 + ((L < 16) ? 0 : 16) + i;
            int ic = krow / 9, r9 = krow % 9;
            int ky = r9 / 3, kx = r9 % 3;
            int p = n0 + (L & 15);
            unsigned short v = 0;
            if (p < P) {
                int y = p / 14, x = p % 14;
                int yy = y + ky - 1, xx = x + kx - 1;
                if (yy >= 0 && yy < 14 && xx >= 0 && xx < 14)
                    v = bfu(Pb[(size_t)ic * 196 + yy * 14 + xx]);
            }
            bt[e] = v;
        }
        // ---- A tile (256 rows x 32 K of Wc) -> LDS ----
#if __has_builtin(__builtin_amdgcn_tensor_load_to_lds)
        if (wave == 0) {
            unsigned long long ga = (unsigned long long)(const void*)Wc + (unsigned long long)k0 * 2ull;
            u32x4 g0;
            g0[0] = 1u;                                       // count=1, user descriptor
            g0[1] = (unsigned)(size_t)(void*)&at[0];          // lds_addr (low 32 bits of flat = LDS offset)
            g0[2] = (unsigned)(ga & 0xffffffffu);             // global_addr[31:0]
            g0[3] = (unsigned)((ga >> 32) & 0x01ffffffu) | 0x80000000u;  // addr[56:32] | type=2
            // data_size=2B; tensor_dim0=4608; tensor_dim1=256; tile_dim0=32; tile_dim1=256; stride0=4608
            i32x8 g1 = { 0x00010000, 0x12000000, 0x01000000, 0x00200000, 0x00000100, 4608, 0, 0 };
            i32x4 gz4 = { 0, 0, 0, 0 };
#if defined(__clang_major__) && (__clang_major__ >= 23)
            i32x8 gz8 = { 0, 0, 0, 0, 0, 0, 0, 0 };
            __builtin_amdgcn_tensor_load_to_lds(g0, g1, gz4, gz4, gz8, 0);
#else
            __builtin_amdgcn_tensor_load_to_lds(g0, g1, gz4, gz4, 0);
#endif
            __builtin_amdgcn_s_wait_tensorcnt(0);
        }
#else
        // fallback: cooperative vector copy, one row per thread
        {
            const __bf16* src = Wc + (size_t)tid * KC + k0;
#pragma unroll
            for (int j = 0; j < 4; j++)
                *(bf16x8*)(void*)&at[tid * 32 + j * 8] = *(const bf16x8*)(src + j * 8);
        }
#endif
        __syncthreads();

        bf16x8 a0lo = *(const bf16x8*)(void*)&at[row0 * 32 + kgrp];
        bf16x8 a0hi = *(const bf16x8*)(void*)&at[row0 * 32 + kgrp + 8];
        bf16x8 a1lo = *(const bf16x8*)(void*)&at[row1 * 32 + kgrp];
        bf16x8 a1hi = *(const bf16x8*)(void*)&at[row1 * 32 + kgrp + 8];
        bf16x16 a0 = __builtin_shufflevector(a0lo, a0hi, 0,1,2,3,4,5,6,7,8,9,10,11,12,13,14,15);
        bf16x16 a1 = __builtin_shufflevector(a1lo, a1hi, 0,1,2,3,4,5,6,7,8,9,10,11,12,13,14,15);
        bf16x16 b  = *(const bf16x16*)(void*)&bt[lane * 16];

        acc0 = __builtin_amdgcn_wmma_f32_16x16x32_bf16(false, a0, false, b, (short)0, acc0, false, false);
        acc1 = __builtin_amdgcn_wmma_f32_16x16x32_bf16(false, a1, false, b, (short)0, acc1, false, false);
        __syncthreads();
    }

    int col = lane & 15, rb = (lane < 16) ? 0 : 8;
    int p = n0 + col;
    if (p < P) {
#pragma unroll
        for (int r = 0; r < 8; r++) {
            int oc0 = wave * 16 + rb + r;
            float v0 = fmaxf(acc0[r] + bconv[oc0], 0.f) * (1.0f / 196.0f);
            atomicAdd(&zmean[box * 256 + oc0], v0);
            int oc1 = (wave + 8) * 16 + rb + r;
            float v1 = fmaxf(acc1[r] + bconv[oc1], 0.f) * (1.0f / 196.0f);
            atomicAdd(&zmean[box * 256 + oc1], v1);
        }
    }
}

// ---------------- small heads ----------------
__global__ void heads_kernel(const float* __restrict__ H, const float* __restrict__ wc,
                             const float* __restrict__ bc, const float* __restrict__ wb,
                             const float* __restrict__ bb, float* __restrict__ out_cls,
                             float* __restrict__ out_reg) {
    int id = blockIdx.x * blockDim.x + threadIdx.x;
    if (id >= 512 * 81) return;
    int m = id / 81, j = id % 81;
    const float* hr = H + (size_t)m * 1024;
    float sc = 0.f, sb = 0.f;
    for (int k = 0; k < 1024; k++) {
        float hv = hr[k];
        sc += hv * wc[k * 81 + j];
        sb += hv * wb[k * 81 + j];
    }
    out_cls[id] = sc + bc[j];
    out_reg[id] = sb + bb[j];
}

__global__ void softmax_kernel(float* __restrict__ out_cls) {
    int m = blockIdx.x * blockDim.x + threadIdx.x;
    if (m >= 512) return;
    float* r = out_cls + m * 81;
    float mx = r[0];
    for (int j = 1; j < 81; j++) mx = fmaxf(mx, r[j]);
    float s = 0.f;
    for (int j = 0; j < 81; j++) { float e = expf(r[j] - mx); r[j] = e; s += e; }
    float inv = 1.f / s;
    for (int j = 0; j < 81; j++) r[j] *= inv;
}

__global__ void mask_head_kernel(const float* __restrict__ zmean, const float* __restrict__ wm,
                                 const float* __restrict__ bm, float* __restrict__ out) {
    int id = blockIdx.x * blockDim.x + threadIdx.x;
    if (id >= 512 * 81) return;
    int m = id / 81, j = id % 81;
    const float* zr = zmean + m * 256;
    float s = 0.f;
    for (int k = 0; k < 256; k++) s += zr[k] * wm[k * 81 + j];
    s += bm[j];
    out[id] = 1.f / (1.f + expf(-s));
}

extern "C" void kernel_launch(void* const* d_in, const int* in_sizes, int n_in,
                              void* d_out, int out_size, void* d_ws, size_t ws_size,
                              hipStream_t stream) {
    (void)in_sizes; (void)n_in; (void)out_size; (void)ws_size;
    const float* p2     = (const float*)d_in[0];
    const float* p3     = (const float*)d_in[1];
    const float* p4     = (const float*)d_in[2];
    const float* p5     = (const float*)d_in[3];
    const float* boxes  = (const float*)d_in[4];
    const int*   bidx   = (const int*)d_in[5];
    const float* w_fc1  = (const float*)d_in[6];
    const float* b_fc1  = (const float*)d_in[7];
    const float* w_cls  = (const float*)d_in[8];
    const float* b_cls  = (const float*)d_in[9];
    const float* w_box  = (const float*)d_in[10];
    const float* b_box  = (const float*)d_in[11];
    const float* w_conv = (const float*)d_in[12];
    const float* b_conv = (const float*)d_in[13];
    const float* w_mfc  = (const float*)d_in[14];
    const float* b_mfc  = (const float*)d_in[15];
    const int*   img_h  = (const int*)d_in[16];
    const int*   img_w  = (const int*)d_in[17];

    // workspace layout (256B aligned)
    char* ws = (char*)d_ws;
    size_t off = 0;
    auto carve = [&](size_t bytes) { size_t r = off; off = (off + bytes + 255) & ~(size_t)255; return r; };
    __bf16* wfc1b   = (__bf16*)(ws + carve((size_t)100352 * 1024 * 2)); // 205.5 MB (swizzled)
    __bf16* pooledb = (__bf16*)(ws + carve((size_t)512 * 100352 * 2));  // 102.8 MB
    float*  Hbuf    = (float*) (ws + carve((size_t)512 * 1024 * 4));    //   2.1 MB
    __bf16* wconvb  = (__bf16*)(ws + carve((size_t)256 * 4608 * 2));    //   2.4 MB
    float*  zmean   = (float*) (ws + carve((size_t)512 * 256 * 4));     //   0.5 MB

    float* out_cls  = (float*)d_out;            // [512,81]
    float* out_reg  = out_cls + 512 * 81;       // [512,81]
    float* out_mask = out_cls + 2 * 512 * 81;   // [512,81]

    // 1) one-time weight conversions (FC1 swizzled to fragment-major)
    {
        long long n1 = (long long)100352 * 1024;
        swizzleB_fc1_kernel<<<(unsigned)((n1 + 255) / 256), 256, 0, stream>>>(w_fc1, wfc1b);
        long long n2 = (long long)256 * 4608;
        cvt_bf16_kernel<<<(unsigned)((n2 + 255) / 256), 256, 0, stream>>>(w_conv, wconvb, n2);
    }
    // 2) zero the atomic accumulator
    zero_kernel<<<512, 256, 0, stream>>>(zmean, 512 * 256);

    // 3) FPN-level + RoIAlign -> pooled bf16
    {
        long long tot = (long long)512 * 512 * 196;
        roi_pool_kernel<<<(unsigned)((tot + 255) / 256), 256, 0, stream>>>(
            p2, p3, p4, p5, boxes, bidx, img_h, img_w, pooledb);
    }

    // 4) FC1 WMMA GEMM + bias + ReLU -> Hbuf  (512 wave-jobs, 8 waves/block)
    fc1_wmma_kernel<<<64, 256, 0, stream>>>(pooledb, wfc1b, b_fc1, Hbuf);

    // 5) cls/box heads + softmax (cls in-place in d_out)
    heads_kernel<<<(512 * 81 + 255) / 256, 256, 0, stream>>>(Hbuf, w_cls, b_cls, w_box, b_box,
                                                             out_cls, out_reg);
    softmax_kernel<<<2, 256, 0, stream>>>(out_cls);

    // 6) conv-as-GEMM WMMA with TDM weight staging, fused bias/ReLU/mean -> zmean
    conv_wmma_kernel<<<512 * 13, 256, 0, stream>>>(wconvb, pooledb, b_conv, zmean);

    // 7) mask head
    mask_head_kernel<<<(512 * 81 + 255) / 256, 256, 0, stream>>>(zmean, w_mfc, b_mfc, out_mask);
}
